// MolecularVAE_30279519437403
// MI455X (gfx1250) — matrix operand order repaired and checked by reading, not compiled, not run on hardware
//
#include <hip/hip_runtime.h>
#include <hip/hip_bf16.h>

// Problem dimensions (compile-time)
#define Bsz 512
#define Tsz 128
#define Vv  64
#define Ee  256
#define Hh  512
#define Ll  128

typedef __attribute__((ext_vector_type(16))) __bf16 v16bf;
typedef __attribute__((ext_vector_type(8)))  float  v8f;

union V16U { v16bf v; unsigned u[8]; };

// Async direct-to-LDS staging if the toolchain exposes the gfx1250 builtins.
#if __has_builtin(__builtin_amdgcn_global_load_async_to_lds_b128) && \
    __has_builtin(__builtin_amdgcn_s_wait_asynccnt)
#define USE_ASYNC_LDS 1
typedef __attribute__((ext_vector_type(4))) int v4i_;
typedef __attribute__((address_space(1))) v4i_* as1_v4i;
typedef __attribute__((address_space(3))) v4i_* as3_v4i;
#else
#define USE_ASYNC_LDS 0
#endif

__device__ __forceinline__ unsigned ld2(const __bf16* p) {
  return *reinterpret_cast<const unsigned*>(p);
}

// A fragment (16x32 bf16 tile), r = A + row*lda + half*8 + k0:
// VGPR v in 0..3 : K = half*8 + 2v ; v in 4..7 : K = 16 + half*8 + 2(v-4)
__device__ __forceinline__ v16bf load_a_frag(const __bf16* r) {
  V16U a;
#pragma unroll
  for (int v = 0; v < 4; ++v) {
    a.u[v]     = ld2(r + 2 * v);
    a.u[4 + v] = ld2(r + 16 + 2 * v);
  }
  return a.v;
}

// B operand tile for D = A * W^T where W is [N,K] row-major (ld = ldw):
// lane idx -> column N = n0+idx; elems 0..7: K = half*8+s, elems 8..15: K = 16+half*8+(s-8).
__device__ __forceinline__ v16bf load_bt_tile(const __bf16* __restrict__ W, int ldw,
                                              int n0, int k0, int idx, int half) {
  V16U b;
  const __bf16* r = W + (size_t)(n0 + idx) * ldw + k0 + half * 8;
#pragma unroll
  for (int v = 0; v < 4; ++v) { b.u[v] = ld2(r + 2 * v); b.u[4 + v] = ld2(r + 16 + 2 * v); }
  return b.v;
}

// Load a pre-staged 32-byte per-lane fragment from LDS (2x ds b128).
__device__ __forceinline__ v16bf ld_frag_lds(const char* p) {
  V16U r;
  ((uint4*)r.u)[0] = ((const uint4*)p)[0];
  ((uint4*)r.u)[1] = ((const uint4*)p)[1];
  return r.v;
}

__device__ __forceinline__ v8f wmma_bf16(v16bf a, v16bf b, v8f c) {
  return __builtin_amdgcn_wmma_f32_16x16x32_bf16(false, a, false, b, (short)0, c, false, false);
}

__device__ __forceinline__ float sigmoidf_(float x) {
  return 1.0f / (1.0f + __expf(-x));
}
// exp-based tanh: stays on the transcendental pipe (no libm precise path)
__device__ __forceinline__ float tanhf_(float x) {
  float e = __expf(-2.0f * x);
  return (1.0f - e) / (1.0f + e);
}

// Stage 16 bytes of the weight stream into LDS (one thread's share of a tile).
__device__ __forceinline__ void stage16(const __bf16* src, char* dst) {
#if USE_ASYNC_LDS
  __builtin_amdgcn_global_load_async_to_lds_b128(
      (as1_v4i)src, (as3_v4i)dst, 0, 0);
#else
  *(uint4*)dst = *(const uint4*)src;
#endif
}
__device__ __forceinline__ void stage_fence() {
#if USE_ASYNC_LDS
  __builtin_amdgcn_s_wait_asynccnt(0);
#endif
}

// ---------------------------------------------------------------------------
// Weight pack: dst (bf16, rows x (colsA+colsB)) = [A | B] converted to bf16.
__global__ void pack_concat_kernel(__bf16* __restrict__ dst,
                                   const float* __restrict__ A, int colsA,
                                   const float* __restrict__ Bm, int colsB, int n) {
  int i = blockIdx.x * blockDim.x + threadIdx.x;
  if (i >= n) return;
  int ld = colsA + colsB;
  int r = i / ld;
  int cmod = i - r * ld;
  float v = (cmod < colsA) ? A[(size_t)r * colsA + cmod]
                           : Bm[(size_t)r * colsB + (cmod - colsA)];
  dst[i] = (__bf16)v;
}

// Embedding gather: dst[t,b,e] (bf16, [T,B,E]) = emb[tok[b,t], e]
__global__ void gather_embed_kernel(__bf16* __restrict__ dst,
                                    const float* __restrict__ emb,
                                    const int* __restrict__ tok, int n) {
  int i = blockIdx.x * blockDim.x + threadIdx.x;
  if (i >= n) return;
  int e = i & (Ee - 1);
  int r = i >> 8;           // / E (E=256)
  int t = r >> 9;           // / B (B=512)
  int b = r & (Bsz - 1);
  int id = tok[b * Tsz + t];
  dst[i] = (__bf16)emb[(size_t)id * Ee + e];
}

// ---------------------------------------------------------------------------
// Per-region inner loop: A row pointer is hoisted, inner stride is constant;
// staging of tile kt+1 (async direct-to-LDS when available) overlaps the 4
// WMMAs on tile kt; one barrier per K-tile swaps the double buffer.
struct StageCtx {
  const __bf16* wrow;  // this thread's 16B share of the weight stream (k=0)
  char* sdst0;         // LDS destination, buffer 0
  char* sdst1;         // LDS destination, buffer 1
  int nk;              // total K-tiles
};

__device__ __forceinline__ void gemm_region(
    const __bf16* arow, int ktBegin, int ktEnd, const StageCtx& st,
    const char* bbBase, v8f& acc0, v8f& acc1, v8f& acc2, v8f& acc3) {
  const __bf16* ar = arow;
  for (int kt = ktBegin; kt < ktEnd; ++kt, ar += 32) {
    int ktN = kt + 1;
    if (ktN < st.nk) {
      stage16(st.wrow + ((size_t)ktN << 5), (ktN & 1) ? st.sdst1 : st.sdst0);
    }
    v16bf a = load_a_frag(ar);
    const char* bb = bbBase + ((kt & 1) << 12);   // 4KB per buffer
    v16bf b0 = ld_frag_lds(bb);
    v16bf b1 = ld_frag_lds(bb + 1024);
    v16bf b2 = ld_frag_lds(bb + 2048);
    v16bf b3 = ld_frag_lds(bb + 3072);
    acc0 = wmma_bf16(a, b0, acc0);
    acc1 = wmma_bf16(a, b1, acc1);
    acc2 = wmma_bf16(a, b2, acc2);
    acc3 = wmma_bf16(a, b3, acc3);
    stage_fence();
    __syncthreads();
  }
}

// One LSTM timestep, fused:  G = [X0 | X1 | Hin] @ Wcat^T + bias, 4 gates,
// c' = sig(f)*c + sig(i)*tanh(g), h' = sig(o)*tanh(c').
// All 8 waves in a block share gate-column tile nT and cooperatively stage the
// 4 gate B-fragments (4KB) in double-buffered LDS (8x less L2 weight traffic);
// each wave owns a distinct row tile mT.
// Grid: exactly 128 blocks x 256 threads = 1024 waves = (B/16)*(H/16).
__global__ __launch_bounds__(256) void lstm_step_kernel(
    const __bf16* __restrict__ X0, int ldx0, int cols0,
    const __bf16* __restrict__ X1, int ldx1, int cols1,
    const __bf16* __restrict__ Hin,
    const __bf16* __restrict__ Wcat, int ldw,
    const float* __restrict__ bias,
    float* __restrict__ c,
    __bf16* __restrict__ h_out) {
  // [buf][gate][lane*32B]  -> 2 * 4 * 1024 = 8KB
  __shared__ __align__(16) char lds[2][4][32 * 32];

  int wib  = threadIdx.x >> 5;              // wave in block 0..7
  int lane = threadIdx.x & 31;
  int idx = lane & 15, half = lane >> 4;
  int nT = blockIdx.x >> 2;                 // 0..31 (shared by block)
  int mT = ((blockIdx.x & 3) << 3) + wib;   // 0..31
  int m0 = mT << 4, n0 = nT << 4;
  int nk0 = cols0 >> 5, nk1 = cols1 >> 5, nkH = Hh >> 5;
  int nk = nk0 + nk1 + nkH;

  // Cooperative stage role: 256 threads move 4KB (128 fragments x 32B, 16B each)
  int f    = threadIdx.x >> 1;              // fragment 0..127
  int part = threadIdx.x & 1;
  int sg   = f >> 5;                        // gate 0..3
  int sl   = f & 31;                        // lane slot 0..31
  int sidx = sl & 15, shalf = sl >> 4;
  StageCtx st;
  st.wrow  = Wcat + (size_t)(n0 + sg * Hh + sidx) * ldw + shalf * 8 + part * 16;
  st.sdst0 = &lds[0][sg][sl * 32 + part * 16];
  st.sdst1 = &lds[1][sg][sl * 32 + part * 16];
  st.nk    = nk;

  // Hoisted per-region A row pointers (regions are multiples of 32 columns)
  const __bf16* a0 = X0 + (size_t)(m0 + idx) * ldx0 + half * 8;
  const __bf16* a1 = (cols1 ? X1 : X0) + (size_t)(m0 + idx) * (cols1 ? ldx1 : ldx0) + half * 8;
  const __bf16* aH = Hin + (size_t)(m0 + idx) * Hh + half * 8;

  // Prologue: stage k-tile 0
  stage16(st.wrow, st.sdst0);
  stage_fence();
  __syncthreads();

  v8f acc0 = {}, acc1 = {}, acc2 = {}, acc3 = {};
  const char* bbBase = &lds[0][0][lane * 32];
  gemm_region(a0, 0,         nk0,       st, bbBase, acc0, acc1, acc2, acc3);
  gemm_region(a1, nk0,       nk0 + nk1, st, bbBase, acc0, acc1, acc2, acc3);
  gemm_region(aH, nk0 + nk1, nk,        st, bbBase, acc0, acc1, acc2, acc3);

  int n = n0 + idx;
  float bi = bias[n], bff = bias[n + Hh], bg = bias[n + 2 * Hh], bo = bias[n + 3 * Hh];
#pragma unroll
  for (int r = 0; r < 8; ++r) {
    int m = m0 + r + half * 8;
    size_t off = (size_t)m * Hh + n;
    float ig = sigmoidf_(acc0[r] + bi);
    float fg = sigmoidf_(acc1[r] + bff);
    float gg = tanhf_(acc2[r] + bg);
    float og = sigmoidf_(acc3[r] + bo);
    float cc = fg * c[off] + ig * gg;
    c[off] = cc;
    h_out[off] = (__bf16)(og * tanhf_(cc));
  }
}

// ---------------------------------------------------------------------------
// Fused latent head: hcat=[hF|hB] (K=1024); mu = hcat@Wmu^T+bmu,
// lv = hcat@Wlv^T+blv, z = mu + eps*exp(0.5*lv).
// Grid: (B/16)*(L/16)=256 waves -> 32 blocks x 256 threads.
__global__ __launch_bounds__(256) void latent_kernel(
    const __bf16* __restrict__ hF, const __bf16* __restrict__ hB,
    const __bf16* __restrict__ Wmu, const __bf16* __restrict__ Wlv,
    const float* __restrict__ bmu, const float* __restrict__ blv,
    const float* __restrict__ eps,
    float* __restrict__ out_mu, float* __restrict__ out_lv,
    __bf16* __restrict__ z) {
  int wave = (blockIdx.x * blockDim.x + threadIdx.x) >> 5;
  int mT = wave >> 3;       // 0..31
  int nT = wave & 7;        // 0..7
  int m0 = mT << 4, n0 = nT << 4;
  int lane = threadIdx.x & 31;
  int idx = lane & 15, half = lane >> 4;
  const __bf16* aF = hF + (size_t)(m0 + idx) * Hh + half * 8;
  const __bf16* aB = hB + (size_t)(m0 + idx) * Hh + half * 8;
  v8f accm = {}, accl = {};
  for (int k0 = 0; k0 < 2 * Hh; k0 += 32) {
    const __bf16* ar = (k0 < Hh) ? (aF + k0) : (aB + (k0 - Hh));
    v16bf a  = load_a_frag(ar);
    v16bf bm = load_bt_tile(Wmu, 2 * Hh, n0, k0, idx, half);
    v16bf bl = load_bt_tile(Wlv, 2 * Hh, n0, k0, idx, half);
    accm = wmma_bf16(a, bm, accm);
    accl = wmma_bf16(a, bl, accl);
  }
  int n = n0 + idx;
  float bmn = bmu[n], bln = blv[n];
#pragma unroll
  for (int r = 0; r < 8; ++r) {
    int m = m0 + r + half * 8;
    size_t o = (size_t)m * Ll + n;
    float mu = accm[r] + bmn;
    float lv = accl[r] + bln;
    out_mu[o] = mu;
    out_lv[o] = lv;
    z[o] = (__bf16)(mu + eps[o] * __expf(0.5f * lv));
  }
}

// Decoder init: hd = z @ W_di^T + b_di   (K=L=128)
// Grid: (B/16)*(H/16)=1024 waves -> 128 blocks x 256 threads.
__global__ __launch_bounds__(256) void decinit_kernel(
    const __bf16* __restrict__ z, const __bf16* __restrict__ Wdi,
    const float* __restrict__ bdi, __bf16* __restrict__ hd) {
  int wave = (blockIdx.x * blockDim.x + threadIdx.x) >> 5;
  int mT = wave >> 5;
  int nT = wave & 31;
  int m0 = mT << 4, n0 = nT << 4;
  int lane = threadIdx.x & 31;
  int idx = lane & 15, half = lane >> 4;
  const __bf16* ar = z + (size_t)(m0 + idx) * Ll + half * 8;
  v8f acc = {};
  for (int k0 = 0; k0 < Ll; k0 += 32) {
    v16bf a = load_a_frag(ar + k0);
    v16bf b = load_bt_tile(Wdi, Ll, n0, k0, idx, half);
    acc = wmma_bf16(a, b, acc);
  }
  int n = n0 + idx;
  float bb = bdi[n];
#pragma unroll
  for (int r = 0; r < 8; ++r) {
    int m = m0 + r + half * 8;
    hd[(size_t)m * Hh + n] = (__bf16)(acc[r] + bb);
  }
}

// Output projection: recon[b,t,v] = hs[t,b,:] @ W_out[v,:] + b_out[v]
// hs flat [T*B, H]. One wave owns a 16-row tile and computes ALL 4 V-tiles,
// reusing the A fragment across 4 WMMAs per K-tile.
// Grid: T*B/16 = 4096 waves -> 512 blocks x 256 threads.
__global__ __launch_bounds__(256) void recon_kernel(
    const __bf16* __restrict__ hs, const __bf16* __restrict__ Wout,
    const float* __restrict__ bout, float* __restrict__ out) {
  int wave = (blockIdx.x * blockDim.x + threadIdx.x) >> 5;
  int m0 = wave << 4;
  int lane = threadIdx.x & 31;
  int idx = lane & 15, half = lane >> 4;
  const __bf16* ar = hs + (size_t)(m0 + idx) * Hh + half * 8;
  v8f acc0 = {}, acc1 = {}, acc2 = {}, acc3 = {};
  for (int k0 = 0; k0 < Hh; k0 += 32) {
    v16bf a  = load_a_frag(ar + k0);
    v16bf b0 = load_bt_tile(Wout, Hh, 0,  k0, idx, half);
    v16bf b1 = load_bt_tile(Wout, Hh, 16, k0, idx, half);
    v16bf b2 = load_bt_tile(Wout, Hh, 32, k0, idx, half);
    v16bf b3 = load_bt_tile(Wout, Hh, 48, k0, idx, half);
    acc0 = wmma_bf16(a, b0, acc0);
    acc1 = wmma_bf16(a, b1, acc1);
    acc2 = wmma_bf16(a, b2, acc2);
    acc3 = wmma_bf16(a, b3, acc3);
  }
  v8f accs[4] = {acc0, acc1, acc2, acc3};
#pragma unroll
  for (int g = 0; g < 4; ++g) {
    int n = (g << 4) + idx;
    float bb = bout[n];
#pragma unroll
    for (int r = 0; r < 8; ++r) {
      int mg = m0 + r + half * 8;     // row in [T*B]
      int t = mg >> 9;                // / B
      int b = mg & (Bsz - 1);
      out[(size_t)b * Tsz * Vv + (size_t)t * Vv + n] = accs[g][r] + bb;
    }
  }
}

// ---------------------------------------------------------------------------
extern "C" void kernel_launch(void* const* d_in, const int* in_sizes, int n_in,
                              void* d_out, int out_size, void* d_ws, size_t ws_size,
                              hipStream_t stream) {
  const int*   x      = (const int*)d_in[0];
  const int*   tgt    = (const int*)d_in[1];
  const float* eps    = (const float*)d_in[2];
  const float* emb    = (const float*)d_in[3];
  const float* Wih_f  = (const float*)d_in[4];
  const float* Whh_f  = (const float*)d_in[5];
  const float* b_f    = (const float*)d_in[6];
  const float* Wih_b  = (const float*)d_in[7];
  const float* Whh_b  = (const float*)d_in[8];
  const float* b_b    = (const float*)d_in[9];
  const float* W_mu   = (const float*)d_in[10];
  const float* b_mu   = (const float*)d_in[11];
  const float* W_lv   = (const float*)d_in[12];
  const float* b_lv   = (const float*)d_in[13];
  const float* W_di   = (const float*)d_in[14];
  const float* b_di   = (const float*)d_in[15];
  const float* Wih_d  = (const float*)d_in[16];
  const float* Whh_d  = (const float*)d_in[17];
  const float* b_d    = (const float*)d_in[18];
  const float* W_out  = (const float*)d_in[19];
  const float* b_out  = (const float*)d_in[20];

  float* out_recon = (float*)d_out;                              // [B,T,V]
  float* out_mu    = out_recon + (size_t)Bsz * Tsz * Vv;         // [B,L]
  float* out_lv    = out_mu + (size_t)Bsz * Ll;                  // [B,L]

  // Workspace bump allocator (256B aligned)
  char* p = (char*)d_ws;
  auto alloc = [&](size_t bytes) -> void* {
    void* r = (void*)p;
    p += (bytes + 255) & ~(size_t)255;
    return r;
  };
  __bf16* ex      = (__bf16*)alloc((size_t)Tsz * Bsz * Ee * 2);        // [T,B,E]
  __bf16* te      = (__bf16*)alloc((size_t)Tsz * Bsz * Ee * 2);        // [T,B,E]
  __bf16* Wcat_f  = (__bf16*)alloc((size_t)4 * Hh * (Ee + Hh) * 2);    // [4H, E+H]
  __bf16* Wcat_b  = (__bf16*)alloc((size_t)4 * Hh * (Ee + Hh) * 2);
  __bf16* Wcat_d  = (__bf16*)alloc((size_t)4 * Hh * (Ee + Ll + Hh) * 2);
  __bf16* Wmu_bf  = (__bf16*)alloc((size_t)Ll * 2 * Hh * 2);
  __bf16* Wlv_bf  = (__bf16*)alloc((size_t)Ll * 2 * Hh * 2);
  __bf16* Wdi_bf  = (__bf16*)alloc((size_t)Hh * Ll * 2);
  __bf16* Wout_bf = (__bf16*)alloc((size_t)Vv * Hh * 2);
  __bf16* hbuf0   = (__bf16*)alloc((size_t)Bsz * Hh * 2);              // fwd double-buffer
  __bf16* hbuf1   = (__bf16*)alloc((size_t)Bsz * Hh * 2);
  __bf16* hbuf2   = (__bf16*)alloc((size_t)Bsz * Hh * 2);              // bwd double-buffer
  __bf16* hbuf3   = (__bf16*)alloc((size_t)Bsz * Hh * 2);
  __bf16* hd_bf   = (__bf16*)alloc((size_t)Bsz * Hh * 2);
  __bf16* z_bf    = (__bf16*)alloc((size_t)Bsz * Ll * 2);
  float*  c_enc   = (float*)alloc((size_t)Bsz * Hh * 4);
  float*  c_dec   = (float*)alloc((size_t)Bsz * Hh * 4);
  __bf16* hs_all  = (__bf16*)alloc((size_t)Tsz * Bsz * Hh * 2);        // [T,B,H]

  // ---- pack weights to bf16 (concat Wih|Whh so each step is one GEMM) ----
  {
    int n = 4 * Hh * (Ee + Hh);
    pack_concat_kernel<<<(n + 255) / 256, 256, 0, stream>>>(Wcat_f, Wih_f, Ee, Whh_f, Hh, n);
    pack_concat_kernel<<<(n + 255) / 256, 256, 0, stream>>>(Wcat_b, Wih_b, Ee, Whh_b, Hh, n);
    int nd = 4 * Hh * (Ee + Ll + Hh);
    pack_concat_kernel<<<(nd + 255) / 256, 256, 0, stream>>>(Wcat_d, Wih_d, Ee + Ll, Whh_d, Hh, nd);
    int nm = Ll * 2 * Hh;
    pack_concat_kernel<<<(nm + 255) / 256, 256, 0, stream>>>(Wmu_bf, W_mu, 2 * Hh, nullptr, 0, nm);
    pack_concat_kernel<<<(nm + 255) / 256, 256, 0, stream>>>(Wlv_bf, W_lv, 2 * Hh, nullptr, 0, nm);
    int ni = Hh * Ll;
    pack_concat_kernel<<<(ni + 255) / 256, 256, 0, stream>>>(Wdi_bf, W_di, Ll, nullptr, 0, ni);
    int no = Vv * Hh;
    pack_concat_kernel<<<(no + 255) / 256, 256, 0, stream>>>(Wout_bf, W_out, Hh, nullptr, 0, no);
  }

  // ---- embedding gathers ----
  {
    int n = Tsz * Bsz * Ee;
    gather_embed_kernel<<<n / 256, 256, 0, stream>>>(ex, emb, x, n);
    gather_embed_kernel<<<n / 256, 256, 0, stream>>>(te, emb, tgt, n);
  }

  const size_t hbytes = (size_t)Bsz * Hh * 2;
  const size_t cbytes = (size_t)Bsz * Hh * 4;
  dim3 stepGrid(128), stepBlk(256);

  // ---- encoder forward scan ----
  (void)hipMemsetAsync(hbuf0, 0, hbytes, stream);
  (void)hipMemsetAsync(c_enc, 0, cbytes, stream);
  const __bf16* hin = hbuf0;
  __bf16* hout = hbuf1;
  for (int t = 0; t < Tsz; ++t) {
    lstm_step_kernel<<<stepGrid, stepBlk, 0, stream>>>(
        ex + (size_t)t * Bsz * Ee, Ee, Ee, nullptr, 0, 0, hin,
        Wcat_f, Ee + Hh, b_f, c_enc, hout);
    const __bf16* tmp = hin; hin = hout; hout = (__bf16*)tmp;
  }
  const __bf16* hF = hin;   // final forward hidden

  // ---- encoder backward scan ----
  (void)hipMemsetAsync(hbuf2, 0, hbytes, stream);
  (void)hipMemsetAsync(c_enc, 0, cbytes, stream);
  hin = hbuf2; hout = hbuf3;
  for (int s = 0; s < Tsz; ++s) {
    int t = Tsz - 1 - s;
    lstm_step_kernel<<<stepGrid, stepBlk, 0, stream>>>(
        ex + (size_t)t * Bsz * Ee, Ee, Ee, nullptr, 0, 0, hin,
        Wcat_b, Ee + Hh, b_b, c_enc, hout);
    const __bf16* tmp = hin; hin = hout; hout = (__bf16*)tmp;
  }
  const __bf16* hBf = hin;  // final backward hidden

  // ---- latent: mu, logvar, z ----
  latent_kernel<<<32, 256, 0, stream>>>(hF, hBf, Wmu_bf, Wlv_bf, b_mu, b_lv,
                                        eps, out_mu, out_lv, z_bf);

  // ---- decoder init hidden ----
  decinit_kernel<<<128, 256, 0, stream>>>(z_bf, Wdi_bf, b_di, hd_bf);
  (void)hipMemsetAsync(c_dec, 0, cbytes, stream);

  // ---- decoder scan (writes h directly into [T,B,H] history) ----
  const __bf16* dh = hd_bf;
  for (int t = 0; t < Tsz; ++t) {
    __bf16* ho = hs_all + (size_t)t * Bsz * Hh;
    lstm_step_kernel<<<stepGrid, stepBlk, 0, stream>>>(
        te + (size_t)t * Bsz * Ee, Ee, Ee, z_bf, Ll, Ll, dh,
        Wcat_d, Ee + Ll + Hh, b_d, c_dec, ho);
    dh = ho;
  }

  // ---- output projection ----
  recon_kernel<<<512, 256, 0, stream>>>(hs_all, Wout_bf, b_out, out_recon);
}
